// HWNN_50483045597670
// MI455X (gfx1250) — matrix operand order repaired and checked by reading, not compiled, not run on hardware
//
#include <hip/hip_runtime.h>
#include <hip/hip_bf16.h>
#include <math.h>

// ---------------------------------------------------------------------------
// HWNN on MI455X (gfx1250).
//  out = (snap0+snap1) * log_softmax(layer2(relu(layer1(features))))
//  layer(x) = (p0 I + p1 T + p2 T^2) diag(d) (p3 I + p4 T^t + p5 (T^t)^2) x W
// 8 big [4096x4096]@[4096x128] GEMMs dominate; Theta+Theta^T are L2-resident.
// GEMMs run on v_wmma_f32_16x16x32_bf16 with hi/lo bf16 split (3 products)
// for ~f32 accuracy with f32 accumulation.
// ---------------------------------------------------------------------------

typedef __bf16 bf16;
typedef bf16  v16bf __attribute__((ext_vector_type(16)));
typedef float v8f   __attribute__((ext_vector_type(8)));

#define NROWS 4096
#define FILT  128

// ---------------------------------------------------------------------------
// C[4096 x 128] = A[4096 x K] (f32 row-major, lda==K) @ B[K x 128]
// B supplied transposed + split: BhT/BlT are [128 x K] bf16, B ~= Bh + Bl.
// One wave per 16-row stripe; 8 accumulator tiles of 16x16 cover 128 cols.
//
// Fragment layouts (A/B use this kernel's contiguous k order; C/D per CDNA5 ISA):
//   A 16x32 bf16 : lane m=l%16; lanes<16 hold K 0..15, lanes>=16
//                  hold K 16..31  -> one contiguous 16-float run.
//   B 32x16 bf16 : lane n=l%16; lanes<16 hold K 0..15, lanes>=16 K 16..31
//                  -> one contiguous 16-bf16 (32B) run per lane.
//   C/D 16x16 f32: vgpr r, lane l -> (M = r + 8*(l>=16), N = l%16).
// ---------------------------------------------------------------------------
__global__ __launch_bounds__(32) void hwnn_wmma_gemm(
    const float* __restrict__ A,
    const bf16*  __restrict__ BhT,
    const bf16*  __restrict__ BlT,
    float* __restrict__ Cmat,
    int K)
{
    const int lane  = threadIdx.x & 31;
    const int hi    = lane >> 4;      // which half-wave
    const int l15   = lane & 15;
    const int mbase = blockIdx.x * 16;

    v8f acc[8] = {};

    const float* Arow = A + (size_t)(mbase + l15) * (size_t)K;

    for (int k0 = 0; k0 < K; k0 += 32) {
        // ---- A fragment (on-the-fly f32 -> bf16 hi/lo split) ----
        const int ka = k0 + hi * 16;
        const float4 a0 = *(const float4*)(Arow + ka);
        const float4 a1 = *(const float4*)(Arow + ka + 4);
        const float4 a2 = *(const float4*)(Arow + ka + 8);
        const float4 a3 = *(const float4*)(Arow + ka + 12);
        float af[16] = {a0.x,a0.y,a0.z,a0.w, a1.x,a1.y,a1.z,a1.w,
                        a2.x,a2.y,a2.z,a2.w, a3.x,a3.y,a3.z,a3.w};
        v16bf Ah, Al;
        #pragma unroll
        for (int i = 0; i < 16; ++i) {
            bf16 h = (bf16)af[i];
            Ah[i] = h;
            Al[i] = (bf16)(af[i] - (float)h);
        }

        // ---- 8 column tiles ----
        const int kb = k0 + hi * 16;
        #pragma unroll
        for (int t = 0; t < 8; ++t) {
            const size_t boff = (size_t)(t * 16 + l15) * (size_t)K + kb;
            v16bf Bh = *(const v16bf*)(BhT + boff);
            v16bf Bl = *(const v16bf*)(BlT + boff);
            acc[t] = __builtin_amdgcn_wmma_f32_16x16x32_bf16(
                         false, Ah, false, Bh, (short)0, acc[t], false, false);
            acc[t] = __builtin_amdgcn_wmma_f32_16x16x32_bf16(
                         false, Ah, false, Bl, (short)0, acc[t], false, false);
            acc[t] = __builtin_amdgcn_wmma_f32_16x16x32_bf16(
                         false, Al, false, Bh, (short)0, acc[t], false, false);
        }
    }

    #pragma unroll
    for (int t = 0; t < 8; ++t) {
        #pragma unroll
        for (int r = 0; r < 8; ++r) {
            Cmat[(size_t)(mbase + hi * 8 + r) * FILT + t * 16 + l15] = acc[t][r];
        }
    }
}

// ---------------------------------------------------------------------------
// Fused elementwise + transpose/convert:
//   Z = act( dvec .* (cA*A + cB*B + cC*C) )       (B,C,dvec optional)
// Optionally writes: Zout (f32 row-major [rows x cols]),
//                    ZoutT (f32 [cols x rows]),
//                    ZhT/ZlT (bf16 hi/lo split, [cols x rows]).
// Null cA => 1.0, null cB/cC => 0.0. 32x32 LDS tile, 32x8 threads.
// ---------------------------------------------------------------------------
__global__ __launch_bounds__(256) void hwnn_prep(
    const float* __restrict__ A, const float* __restrict__ B,
    const float* __restrict__ C,
    const float* cA, const float* cB, const float* cC,
    const float* __restrict__ dvec, int relu,
    float* __restrict__ Zout, float* __restrict__ ZoutT,
    bf16* __restrict__ ZhT, bf16* __restrict__ ZlT,
    int rows, int cols)
{
    __shared__ float tile[32][33];
    const int c  = blockIdx.x * 32 + threadIdx.x;
    const int r0 = blockIdx.y * 32;
    const float ka = cA ? *cA : 1.0f;
    const float kb = cB ? *cB : 0.0f;
    const float kc = cC ? *cC : 0.0f;

    for (int rr = threadIdx.y; rr < 32; rr += 8) {
        const int r = r0 + rr;
        const size_t idx = (size_t)r * (size_t)cols + c;
        float v = ka * A[idx];
        if (B) v += kb * B[idx];
        if (C) v += kc * C[idx];
        if (dvec) v *= dvec[r];
        if (relu) v = fmaxf(v, 0.0f);
        if (Zout) Zout[idx] = v;
        tile[rr][threadIdx.x] = v;
    }
    __syncthreads();

    const int oc0 = blockIdx.x * 32;   // original cols -> output rows
    for (int cc = threadIdx.y; cc < 32; cc += 8) {
        const int tr = oc0 + cc;
        const int tc = r0 + threadIdx.x;
        const float v = tile[threadIdx.x][cc];
        const size_t oidx = (size_t)tr * (size_t)rows + tc;
        if (ZoutT) ZoutT[oidx] = v;
        if (ZhT) {
            bf16 h = (bf16)v;
            ZhT[oidx] = h;
            if (ZlT) ZlT[oidx] = (bf16)(v - (float)h);
        }
    }
}

// ---------------------------------------------------------------------------
// out[row,:] = (sum_s snap[s]) * log_softmax(H[row,:]) ; 128 cols per row.
// ---------------------------------------------------------------------------
__global__ __launch_bounds__(128) void hwnn_logsoftmax(
    const float* __restrict__ H, const float* __restrict__ snap, int nh,
    float* __restrict__ out)
{
    const int row = blockIdx.x;
    const int t   = threadIdx.x;
    const float v = H[(size_t)row * FILT + t];

    __shared__ float red[4];
    float m = v;
    #pragma unroll
    for (int off = 16; off > 0; off >>= 1) m = fmaxf(m, __shfl_xor(m, off, 32));
    if ((t & 31) == 0) red[t >> 5] = m;
    __syncthreads();
    m = fmaxf(fmaxf(red[0], red[1]), fmaxf(red[2], red[3]));
    __syncthreads();

    float s = expf(v - m);
    #pragma unroll
    for (int off = 16; off > 0; off >>= 1) s += __shfl_xor(s, off, 32);
    if ((t & 31) == 0) red[t >> 5] = s;
    __syncthreads();
    s = red[0] + red[1] + red[2] + red[3];

    const float lse = m + logf(s);
    float scale = 0.0f;
    for (int i = 0; i < nh; ++i) scale += snap[i];
    out[(size_t)row * FILT + t] = scale * (v - lse);
}

// ---------------------------------------------------------------------------
extern "C" void kernel_launch(void* const* d_in, const int* in_sizes, int n_in,
                              void* d_out, int out_size, void* d_ws, size_t ws_size,
                              hipStream_t stream)
{
    (void)in_sizes; (void)n_in; (void)out_size; (void)ws_size;

    const float* features = (const float*)d_in[0];   // [4096 x 512]
    const float* Theta    = (const float*)d_in[1];   // [4096 x 4096]
    const float* W1       = (const float*)d_in[2];   // [512 x 128]
    const float* d1       = (const float*)d_in[3];   // [4096]
    const float* par1     = (const float*)d_in[4];   // [6]
    const float* W2       = (const float*)d_in[5];   // [128 x 128]
    const float* d2       = (const float*)d_in[6];   // [4096]
    const float* par2     = (const float*)d_in[7];   // [6]
    const float* snap     = (const float*)d_in[8];   // [2]
    float* out = (float*)d_out;

    // ---- workspace carve-up (~80.3 MB) ----
    char* ws = (char*)d_ws;
    size_t o = 0;
    float* ThetaT = (float*)(ws + o); o += (size_t)NROWS * NROWS * sizeof(float);
    float* fw     = (float*)(ws + o); o += (size_t)NROWS * FILT * sizeof(float);
    float* t1     = (float*)(ws + o); o += (size_t)NROWS * FILT * sizeof(float);
    float* t2     = (float*)(ws + o); o += (size_t)NROWS * FILT * sizeof(float);
    float* ybuf   = (float*)(ws + o); o += (size_t)NROWS * FILT * sizeof(float);
    float* p1b    = (float*)(ws + o); o += (size_t)NROWS * FILT * sizeof(float);
    float* p2b    = (float*)(ws + o); o += (size_t)NROWS * FILT * sizeof(float);
    float* h1b    = (float*)(ws + o); o += (size_t)NROWS * FILT * sizeof(float);
    bf16*  XhT    = (bf16*)(ws + o);  o += (size_t)FILT * NROWS * sizeof(bf16);
    bf16*  XlT    = (bf16*)(ws + o);  o += (size_t)FILT * NROWS * sizeof(bf16);
    bf16*  WTh    = (bf16*)(ws + o);  o += (size_t)FILT * 512 * sizeof(bf16);
    bf16*  WTl    = (bf16*)(ws + o);  o += (size_t)FILT * 512 * sizeof(bf16);

    const dim3 pb(32, 8);
    auto prep = [&](const float* A, const float* B, const float* C,
                    const float* cA, const float* cB, const float* cC,
                    const float* dv, int relu, float* Z, float* ZT,
                    bf16* Zh, bf16* Zl, int rows, int cols) {
        dim3 pg(cols / 32, rows / 32);
        hwnn_prep<<<pg, pb, 0, stream>>>(A, B, C, cA, cB, cC, dv, relu,
                                         Z, ZT, Zh, Zl, rows, cols);
    };
    auto gemm = [&](const float* A, const bf16* Bh, const bf16* Bl,
                    float* C, int K) {
        hwnn_wmma_gemm<<<dim3(NROWS / 16), dim3(32), 0, stream>>>(A, Bh, Bl, C, K);
    };

    // Theta^T (f32) once: subsequent T^t GEMMs read contiguous rows, and
    // Theta+Theta^T (128 MB) stay resident in the 192 MB L2.
    prep(Theta, nullptr, nullptr, nullptr, nullptr, nullptr, nullptr, 0,
         nullptr, ThetaT, nullptr, nullptr, NROWS, NROWS);

    // ================= layer 1 =================
    prep(W1, nullptr, nullptr, nullptr, nullptr, nullptr, nullptr, 0,
         nullptr, nullptr, WTh, WTl, 512, FILT);
    gemm(features, WTh, WTl, fw, 512);                       // fw = x W1
    prep(fw, nullptr, nullptr, nullptr, nullptr, nullptr, nullptr, 0,
         nullptr, nullptr, XhT, XlT, NROWS, FILT);
    gemm(ThetaT, XhT, XlT, t1, NROWS);                       // t1 = T^t fw
    prep(t1, nullptr, nullptr, nullptr, nullptr, nullptr, nullptr, 0,
         nullptr, nullptr, XhT, XlT, NROWS, FILT);
    gemm(ThetaT, XhT, XlT, t2, NROWS);                       // t2 = T^t t1
    prep(fw, t1, t2, par1 + 3, par1 + 4, par1 + 5, d1, 0,    // y = d1.*(...)
         ybuf, nullptr, XhT, XlT, NROWS, FILT);
    gemm(Theta, XhT, XlT, p1b, NROWS);                       // p1 = T y
    prep(p1b, nullptr, nullptr, nullptr, nullptr, nullptr, nullptr, 0,
         nullptr, nullptr, XhT, XlT, NROWS, FILT);
    gemm(Theta, XhT, XlT, p2b, NROWS);                       // p2 = T p1
    prep(ybuf, p1b, p2b, par1 + 0, par1 + 1, par1 + 2, nullptr, 1,
         h1b, nullptr, nullptr, nullptr, NROWS, FILT);       // h1 = relu(...)

    // ================= layer 2 =================
    prep(W2, nullptr, nullptr, nullptr, nullptr, nullptr, nullptr, 0,
         nullptr, nullptr, WTh, WTl, FILT, FILT);
    gemm(h1b, WTh, WTl, fw, FILT);                           // fw = h1 W2
    prep(fw, nullptr, nullptr, nullptr, nullptr, nullptr, nullptr, 0,
         nullptr, nullptr, XhT, XlT, NROWS, FILT);
    gemm(ThetaT, XhT, XlT, t1, NROWS);
    prep(t1, nullptr, nullptr, nullptr, nullptr, nullptr, nullptr, 0,
         nullptr, nullptr, XhT, XlT, NROWS, FILT);
    gemm(ThetaT, XhT, XlT, t2, NROWS);
    prep(fw, t1, t2, par2 + 3, par2 + 4, par2 + 5, d2, 0,
         ybuf, nullptr, XhT, XlT, NROWS, FILT);
    gemm(Theta, XhT, XlT, p1b, NROWS);
    prep(p1b, nullptr, nullptr, nullptr, nullptr, nullptr, nullptr, 0,
         nullptr, nullptr, XhT, XlT, NROWS, FILT);
    gemm(Theta, XhT, XlT, p2b, NROWS);
    prep(ybuf, p1b, p2b, par2 + 0, par2 + 1, par2 + 2, nullptr, 0,
         h1b, nullptr, nullptr, nullptr, NROWS, FILT);       // h2 (reuse h1b)

    // Snapshot loop collapses: acc = (snap0+snap1) * log_softmax(h2).
    hwnn_logsoftmax<<<dim3(NROWS), dim3(128), 0, stream>>>(h1b, snap, 2, out);
}
